// TokenFusion_2757369004442
// MI455X (gfx1250) — matrix operand order repaired and checked by reading, not compiled, not run on hardware
//
#include <hip/hip_runtime.h>

typedef __bf16 bf16_t;
typedef __attribute__((ext_vector_type(16))) __bf16 v16bf;
typedef __attribute__((ext_vector_type(8)))  __bf16 v8bf;
typedef __attribute__((ext_vector_type(8)))  float  v8f;
typedef __attribute__((ext_vector_type(4)))  unsigned int v4u;
typedef __attribute__((ext_vector_type(8)))  int v8i_;
typedef __attribute__((ext_vector_type(4)))  int v4i_;

#define B_   512
#define T_   49
#define D_   256
#define H_   8
#define DH_  32
#define M_   1024
#define L_   4
#define OUT_ 151
#define NT_  (B_*T_)      /* 25088 tokens, divisible by 256 */
#define THR_ 0.02f

// --- CDNA5 TDM availability / toolchain dispatch ---------------------------
#if defined(__gfx1250__) && __has_builtin(__builtin_amdgcn_tensor_load_to_lds) \
                         && __has_builtin(__builtin_amdgcn_s_wait_tensorcnt)
#define HAVE_TDM 1
#else
#define HAVE_TDM 0
#endif
#if __has_include(<hip/amd_detail/amd_gfx1250_TDM.h>)
#define TDM_6ARG 1   /* therock/clang-23 headers -> 6-arg builtin */
#else
#define TDM_6ARG 0   /* ROCm 7.2 / clang-22 -> 5-arg builtin */
#endif

static __device__ __forceinline__ float gelu_f(float x) {
    const float c0 = 0.7978845608028654f; // sqrt(2/pi)
    return 0.5f * x * (1.f + tanhf(c0 * (x + 0.044715f * x * x * x)));
}

#if HAVE_TDM
// Issue one TDM 2D tile load: 4 rows (n-tiles) x tileElems0 bf16, row stride
// rowStride elems, from gsrc into LDS offset ldsOff.  D# per ISA 08 §8.3/8.4.
static __device__ __forceinline__ void tdm_load_panel(const bf16_t* gsrc,
        unsigned ldsOff, unsigned tileElems0, unsigned rowStride)
{
    unsigned long long ga = (unsigned long long)(uintptr_t)gsrc;
    v4u g0;
    g0.x = 1u;                                        // count=1, user descriptor
    g0.y = ldsOff;                                    // lds_addr (bytes)
    g0.z = (unsigned)(ga & 0xffffffffu);              // global_addr[31:0]
    g0.w = (unsigned)((ga >> 32) & 0x01ffffffu)       // global_addr[56:32]
         | (2u << 30);                                // type = 2 ("image")
    v8i_ g1;
    g1[0] = (int)(1u << 16);                          // data_size = 1 -> 2 bytes
    g1[1] = (int)((rowStride & 0xffffu) << 16);       // tensor_dim0[15:0]
    g1[2] = (int)((rowStride >> 16) | (4u << 16));    // tensor_dim0[31:16]|dim1=4
    g1[3] = (int)((tileElems0 & 0xffffu) << 16);      // dim1 hi=0 | tile_dim0
    g1[4] = (int)4u;                                  // tile_dim1=4, tile_dim2=0
    g1[5] = (int)rowStride;                           // tensor_dim0_stride[31:0]
    g1[6] = 0;                                        // stride0 hi | dim1_stride lo
    g1[7] = 0;
    v4i_ z4 = {0, 0, 0, 0};
#if TDM_6ARG
    v8i_ z8 = {0, 0, 0, 0, 0, 0, 0, 0};
    __builtin_amdgcn_tensor_load_to_lds(g0, g1, z4, z4, z8, 0);
#else
    __builtin_amdgcn_tensor_load_to_lds(g0, g1, z4, z4, 0);
#endif
}
#endif

// ---------------------------------------------------------------------------
// Weight repack: W f32 [K,N] row-major  ->  bf16 WMMA-B tiles.
// Tile (nt,kt) is a 32(K)x16(N) block stored [n:16][k:32]; layout is nt-major
// so a block's whole 4-tile x K panel is a strided-2D region (TDM friendly).
// ---------------------------------------------------------------------------
__global__ __launch_bounds__(256)
void k_pack(const float* __restrict__ W, bf16_t* __restrict__ P,
            int K, int N, int Npad)
{
    size_t i = (size_t)blockIdx.x * 256 + threadIdx.x;
    size_t total = (size_t)Npad * K;
    if (i >= total) return;
    int e    = (int)(i & 511);
    size_t tile = i >> 9;
    int nKT  = K >> 5;
    int kt   = (int)(tile % nKT);
    int nt   = (int)(tile / nKT);
    int n    = e >> 5;
    int k    = e & 31;
    int col  = nt * 16 + n;
    int rowk = kt * 32 + k;
    float v  = (col < N) ? W[(size_t)rowk * N + col] : 0.f;
    P[i] = (bf16_t)v;
}

// ---------------------------------------------------------------------------
// LayerNorm over D=256, one wave per row, bf16 output (GEMM A operand).
// ---------------------------------------------------------------------------
__global__ __launch_bounds__(256)
void k_ln(const float* __restrict__ x, const float* __restrict__ g,
          const float* __restrict__ b, bf16_t* __restrict__ out, int Nr)
{
    int wave = threadIdx.x >> 5, lane = threadIdx.x & 31;
    int row  = blockIdx.x * 8 + wave;
    if (row >= Nr) return;
    const float* xr = x + (size_t)row * D_;
    float v[8];
    float s = 0.f;
#pragma unroll
    for (int i = 0; i < 8; ++i) { v[i] = xr[lane + 32 * i]; s += v[i]; }
#pragma unroll
    for (int off = 16; off > 0; off >>= 1) s += __shfl_xor(s, off, 32);
    float mu = s * (1.f / D_);
    float var = 0.f;
#pragma unroll
    for (int i = 0; i < 8; ++i) { float d = v[i] - mu; var += d * d; }
#pragma unroll
    for (int off = 16; off > 0; off >>= 1) var += __shfl_xor(var, off, 32);
    float inv = rsqrtf(var * (1.f / D_) + 1e-5f);
    bf16_t* orow = out + (size_t)row * D_;
#pragma unroll
    for (int i = 0; i < 8; ++i) {
        int c = lane + 32 * i;
        orow[c] = (bf16_t)((v[i] - mu) * inv * g[c] + b[c]);
    }
}

// ---------------------------------------------------------------------------
// WMMA GEMM:  C[Nr,Ncols] = act(A[Nr,K] @ Bpacked + bias) + resid
//   block = 256 thr = 8 waves; wave computes 32(M) x 64(N): 2 A fragments x
//   4 B fragments -> 8 wmma per 32-wide K step (B reused across both M tiles).
//   B panel for the block (4 n-tiles x <=16 k-tiles, <=64KB) is staged into
//   LDS by wave 0 via TENSOR_LOAD_TO_LDS + s_wait_tensorcnt (TDM), barrier,
//   then all waves read fragments with ds loads.  Fallback: direct global.
//   Epilogue order: bias -> gelu -> residual; stores f32 and/or bf16.
// ---------------------------------------------------------------------------
__global__ __launch_bounds__(256)
void k_gemm(const bf16_t* __restrict__ A, const bf16_t* __restrict__ Bp,
            const float* __restrict__ bias, const float* __restrict__ resid,
            float* __restrict__ Cf, bf16_t* __restrict__ Cb,
            int Nr, int K, int Ncols, int act)
{
    const int lane    = threadIdx.x & 31;
    const int wave    = threadIdx.x >> 5;
    const int half    = lane >> 4;       // lanes 0-15 vs 16-31
    const int ln16    = lane & 15;
    const int nKT     = K >> 5;
    const int ntBase  = blockIdx.y * 4;  // 4 n-tiles per wave / block
    const int rowBase = blockIdx.x * 256 + wave * 32;

    int row0 = rowBase + ln16;
    int row1 = row0 + 16;
    const bf16_t* arow0 = A + (size_t)(row0 < Nr ? row0 : Nr - 1) * K;
    const bf16_t* arow1 = A + (size_t)(row1 < Nr ? row1 : Nr - 1) * K;

#if HAVE_TDM
    extern __shared__ __align__(128) char smem[];
    bf16_t* ldsB = (bf16_t*)smem;
#endif

    v8f acc[2][4] = {};

    for (int kc = 0; kc < nKT; kc += 16) {
        const int kn = (nKT - kc) < 16 ? (nKT - kc) : 16;
#if HAVE_TDM
        if (kc) __syncthreads();            // previous chunk fully consumed
        if (threadIdx.x < 32) {             // wave 0 issues the DMA
            tdm_load_panel(Bp + ((size_t)ntBase * nKT + kc) * 512,
                           (unsigned)(uintptr_t)(void*)ldsB,
                           (unsigned)(kn * 512), (unsigned)(nKT * 512));
            __builtin_amdgcn_s_wait_tensorcnt(0);
        }
        __syncthreads();
#endif
        for (int kt2 = 0; kt2 < kn; ++kt2) {
            const int kt = kc + kt2;
            const int k0 = kt * 32 + half * 16;
            // A fragments (contiguous layout: 16 consecutive k per lane half)
            v8bf a0lo = *(const v8bf*)(arow0 + k0);
            v8bf a0hi = *(const v8bf*)(arow0 + k0 + 8);
            v8bf a1lo = *(const v8bf*)(arow1 + k0);
            v8bf a1hi = *(const v8bf*)(arow1 + k0 + 8);
            __builtin_prefetch(arow0 + k0 + 32, 0, 3);
            __builtin_prefetch(arow1 + k0 + 32, 0, 3);
            v16bf a0, a1;
#pragma unroll
            for (int i = 0; i < 8; ++i) {
                a0[i] = a0lo[i]; a0[i + 8] = a0hi[i];
                a1[i] = a1lo[i]; a1[i + 8] = a1hi[i];
            }
#pragma unroll
            for (int j = 0; j < 4; ++j) {
#if HAVE_TDM
                const bf16_t* bptr = ldsB + (size_t)(j * kn + kt2) * 512
                                          + ln16 * 32 + half * 16;
#else
                const bf16_t* bptr = Bp + ((size_t)(ntBase + j) * nKT + kt) * 512
                                        + ln16 * 32 + half * 16;
#endif
                v16bf bfrag = *(const v16bf*)bptr;
                acc[0][j] = __builtin_amdgcn_wmma_f32_16x16x32_bf16(
                                false, a0, false, bfrag, (short)0, acc[0][j],
                                false, false);
                acc[1][j] = __builtin_amdgcn_wmma_f32_16x16x32_bf16(
                                false, a1, false, bfrag, (short)0, acc[1][j],
                                false, false);
            }
        }
    }

    // D layout: VGPR v -> row v (+8 for lanes 16-31), col = lane&15
#pragma unroll
    for (int j = 0; j < 4; ++j) {
        int c = (ntBase + j) * 16 + ln16;
        if (c >= Ncols) continue;
        float bv = bias ? bias[c] : 0.f;
#pragma unroll
        for (int mm = 0; mm < 2; ++mm) {
#pragma unroll
            for (int v = 0; v < 8; ++v) {
                int r = rowBase + mm * 16 + v + half * 8;
                if (r >= Nr) continue;
                float x = acc[mm][j][v] + bv;
                if (act == 1) x = gelu_f(x);
                size_t o = (size_t)r * Ncols + c;
                if (resid) x += resid[o];
                if (Cf) Cf[o] = x;
                if (Cb) Cb[o] = (bf16_t)x;
            }
        }
    }
}

// ---------------------------------------------------------------------------
// Attention: grid = B*H, block = 64.  K/V staged in LDS, online softmax in
// registers, writes bf16 O (A operand of the O-projection GEMM).
// ---------------------------------------------------------------------------
__global__ __launch_bounds__(64)
void k_attn(const float* __restrict__ qkv, bf16_t* __restrict__ obf)
{
    __shared__ float Ks[T_][DH_];
    __shared__ float Vs[T_][DH_];
    int b = blockIdx.x / H_, h = blockIdx.x % H_;
    const float* base = qkv + (size_t)b * T_ * 768;
    for (int i = threadIdx.x; i < T_ * DH_; i += 64) {
        int t = i >> 5, d = i & 31;
        Ks[t][d] = base[t * 768 + 256 + h * DH_ + d];
        Vs[t][d] = base[t * 768 + 512 + h * DH_ + d];
    }
    __syncthreads();
    int qt = threadIdx.x;
    if (qt >= T_) return;
    float q[DH_];
#pragma unroll
    for (int d = 0; d < DH_; ++d) q[d] = base[qt * 768 + h * DH_ + d];

    float m = -1e30f, sum = 0.f;
    float o[DH_] = {};
    for (int t = 0; t < T_; ++t) {
        float s = 0.f;
#pragma unroll
        for (int d = 0; d < DH_; ++d) s += q[d] * Ks[t][d];
        s *= 0.1767766952966369f;            // 1/sqrt(32)
        if (s > m) {
            float sc = __expf(m - s);
            sum *= sc;
#pragma unroll
            for (int d = 0; d < DH_; ++d) o[d] *= sc;
            m = s;
        }
        float p = __expf(s - m);
        sum += p;
#pragma unroll
        for (int d = 0; d < DH_; ++d) o[d] += p * Vs[t][d];
    }
    float inv = 1.f / sum;
    bf16_t* orow = obf + ((size_t)b * T_ + qt) * D_ + h * DH_;
#pragma unroll
    for (int d = 0; d < DH_; ++d) orow[d] = (bf16_t)(o[d] * inv);
}

// ---------------------------------------------------------------------------
// Score head: logits = h2[.,128] @ w3[128,2] + b3, softmax, keep p[0].
// ---------------------------------------------------------------------------
__global__ __launch_bounds__(256)
void k_score(const float* __restrict__ h2, const float* __restrict__ w3,
             const float* __restrict__ b3, float* __restrict__ scores,
             float* __restrict__ pred, int layer, int s)
{
    int t = blockIdx.x * 256 + threadIdx.x;
    if (t >= NT_) return;
    const float* hr = h2 + (size_t)t * 128;
    float l0 = b3[0], l1 = b3[1];
    for (int k = 0; k < 128; ++k) { float v = hr[k]; l0 += v * w3[2 * k]; l1 += v * w3[2 * k + 1]; }
    float mx = fmaxf(l0, l1);
    float e0 = __expf(l0 - mx), e1 = __expf(l1 - mx);
    float p = e0 / (e0 + e1);
    scores[(size_t)s * NT_ + t] = p;
    int b = t / T_, tt = t % T_;
    pred[(size_t)b * (L_ * 5 * T_) + (size_t)layer * 5 * T_ + s * T_ + tt] = p;
}

// mask semantics (faithful to reference): fm = (s>THR)?1:0 ; fnm = ~mask = -2/-1
__global__ __launch_bounds__(256)
void k_fuse_img(const float* __restrict__ t0, const float* __restrict__ t1,
                const float* __restrict__ scores,
                float* __restrict__ f0, float* __restrict__ f1)
{
    size_t i = (size_t)blockIdx.x * 256 + threadIdx.x;
    if (i >= (size_t)NT_ * D_) return;
    size_t t = i / D_;
    float s0 = scores[t], s1 = scores[(size_t)NT_ + t];
    float fm0 = s0 > THR_ ? 1.f : 0.f, fn0 = s0 > THR_ ? -2.f : -1.f;
    float fm1 = s1 > THR_ ? 1.f : 0.f, fn1 = s1 > THR_ ? -2.f : -1.f;
    float a = t0[i], c = t1[i];
    f0[i] = a * fm0 + c * fn0;
    f1[i] = c * fm1 + a * fn1;
}

__global__ __launch_bounds__(256)
void k_gather(const float* __restrict__ tok, const int* __restrict__ idx,
              bf16_t* __restrict__ out)
{
    size_t i = (size_t)blockIdx.x * 256 + threadIdx.x;
    if (i >= (size_t)NT_ * D_) return;
    size_t bt = i / D_;
    int d = (int)(i % D_);
    int b = (int)(bt / T_), t = (int)(bt % T_);
    int src = idx[b * T_ + t];
    out[i] = (bf16_t)tok[((size_t)b * T_ + src) * D_ + d];
}

__global__ __launch_bounds__(256)
void k_fuse_pt(const float* __restrict__ pt, const float* __restrict__ patch,
               const float* __restrict__ scores, int s, float* __restrict__ out)
{
    size_t i = (size_t)blockIdx.x * 256 + threadIdx.x;
    if (i >= (size_t)NT_ * D_) return;
    size_t t = i / D_;
    float sc = scores[(size_t)s * NT_ + t];
    float fm = sc > THR_ ? 1.f : 0.f, fn = sc > THR_ ? -2.f : -1.f;
    out[i] = pt[i] * fm + patch[i] * fn;
}

__global__ __launch_bounds__(256)
void k_pool(const float* __restrict__ f0, const float* __restrict__ f1,
            const float* __restrict__ f2, const float* __restrict__ f3,
            const float* __restrict__ f4, float* __restrict__ pooled)
{
    int i = blockIdx.x * 256 + threadIdx.x;
    if (i >= B_ * D_) return;
    int b = i / D_, d = i % D_;
    const float* fs[5] = { f0, f1, f2, f3, f4 };
    float m = -1e30f;
    for (int s = 0; s < 5; ++s) {
        const float* p = fs[s] + (size_t)b * T_ * D_ + d;
        for (int t = 0; t < T_; ++t) m = fmaxf(m, p[(size_t)t * D_]);
    }
    pooled[i] = m;
}

// ---------------------------------------------------------------------------
extern "C" void kernel_launch(void* const* d_in, const int* in_sizes, int n_in,
                              void* d_out, int out_size, void* d_ws, size_t ws_size,
                              hipStream_t stream)
{
    (void)in_sizes; (void)n_in; (void)out_size; (void)ws_size;

    const float* img0     = (const float*)d_in[0];
    const float* img1     = (const float*)d_in[1];
    const float* p0       = (const float*)d_in[2];
    const float* p1       = (const float*)d_in[3];
    const float* p2       = (const float*)d_in[4];
    const int*   idx      = (const int*)  d_in[5];
    const float* enc_ln1g = (const float*)d_in[6];
    const float* enc_ln1b = (const float*)d_in[7];
    const float* enc_wqkv = (const float*)d_in[8];
    const float* enc_wo   = (const float*)d_in[9];
    const float* enc_bo   = (const float*)d_in[10];
    const float* enc_ln2g = (const float*)d_in[11];
    const float* enc_ln2b = (const float*)d_in[12];
    const float* enc_w1   = (const float*)d_in[13];
    const float* enc_b1   = (const float*)d_in[14];
    const float* enc_w2   = (const float*)d_in[15];
    const float* enc_b2   = (const float*)d_in[16];
    const float* sc_lng   = (const float*)d_in[17];
    const float* sc_lnb   = (const float*)d_in[18];
    const float* sc_w1    = (const float*)d_in[19];
    const float* sc_b1    = (const float*)d_in[20];
    const float* sc_w2    = (const float*)d_in[21];
    const float* sc_b2    = (const float*)d_in[22];
    const float* sc_w3    = (const float*)d_in[23];
    const float* sc_b3    = (const float*)d_in[24];
    const float* pm_w1    = (const float*)d_in[25];
    const float* pm_b1    = (const float*)d_in[26];
    const float* pm_w2    = (const float*)d_in[27];
    const float* pm_b2    = (const float*)d_in[28];
    const float* pm_w3    = (const float*)d_in[29];
    const float* pm_b3    = (const float*)d_in[30];
    const float* hd_lng   = (const float*)d_in[31];
    const float* hd_lnb   = (const float*)d_in[32];
    const float* hd_w1    = (const float*)d_in[33];
    const float* hd_b1    = (const float*)d_in[34];
    const float* hd_w2    = (const float*)d_in[35];
    const float* hd_b2    = (const float*)d_in[36];

    float* smplx = (float*)d_out;                       // [B, 151]
    float* pred  = (float*)d_out + (size_t)B_ * OUT_;   // [B, L*5*T]

    // ---- workspace bump allocator ------------------------------------------
    char* ws = (char*)d_ws;
    size_t off = 0;
    auto alloc = [&](size_t bytes) -> char* {
        char* p = ws + off;
        off = (off + bytes + 255) & ~(size_t)255;
        return p;
    };

    float* tok[5]; float* feat[5];
    for (int s = 0; s < 5; ++s) tok[s]  = (float*)alloc((size_t)NT_ * D_ * 4);
    for (int s = 0; s < 5; ++s) feat[s] = (float*)alloc((size_t)NT_ * D_ * 4);
    float*  qkvf   = (float*) alloc((size_t)NT_ * 768 * 4);
    bf16_t* hbf    = (bf16_t*)alloc((size_t)NT_ * D_ * 2);
    bf16_t* obf    = (bf16_t*)alloc((size_t)NT_ * D_ * 2);
    bf16_t* xbf1   = (bf16_t*)alloc((size_t)NT_ * M_ * 2);
    bf16_t* xbf2   = (bf16_t*)alloc((size_t)NT_ * M_ * 2);
    float*  h2f    = (float*) alloc((size_t)NT_ * 128 * 4);
    float*  patchA = (float*) alloc((size_t)NT_ * D_ * 4);
    float*  patchB = (float*) alloc((size_t)NT_ * D_ * 4);
    bf16_t* g0bf   = (bf16_t*)alloc((size_t)NT_ * D_ * 2);
    bf16_t* g1bf   = (bf16_t*)alloc((size_t)NT_ * D_ * 2);
    float*  scoresb= (float*) alloc((size_t)5 * NT_ * 4);
    float*  pooled = (float*) alloc((size_t)B_ * D_ * 4);

    // ---- pack all weights into WMMA-B bf16 tile layout ---------------------
    auto packW = [&](const float* W, int K, int N, int Npad) -> bf16_t* {
        bf16_t* P = (bf16_t*)alloc((size_t)Npad * K * 2);
        size_t tot = (size_t)Npad * K;
        k_pack<<<dim3((unsigned)((tot + 255) / 256)), 256, 0, stream>>>(W, P, K, N, Npad);
        return P;
    };

    bf16_t *pk_qkv[2][L_], *pk_wo[2][L_], *pk_f1[2][L_], *pk_f2[2][L_];
    for (int e = 0; e < 2; ++e)
        for (int i = 0; i < L_; ++i) {
            size_t li = (size_t)(e * L_ + i);
            pk_qkv[e][i] = packW(enc_wqkv + li * D_ * 768, D_, 768, 768);
            pk_wo [e][i] = packW(enc_wo   + li * D_ * D_,  D_, D_,  D_);
            pk_f1 [e][i] = packW(enc_w1   + li * D_ * M_,  D_, M_,  M_);
            pk_f2 [e][i] = packW(enc_w2   + li * M_ * D_,  M_, D_,  D_);
        }
    bf16_t *pk_s1[L_], *pk_s2[L_], *pk_p1[L_], *pk_p2[L_], *pk_p3[L_];
    for (int i = 0; i < L_; ++i) {
        pk_s1[i] = packW(sc_w1 + (size_t)i * D_ * D_,  D_, D_,  D_);
        pk_s2[i] = packW(sc_w2 + (size_t)i * D_ * 128, D_, 128, 128);
        pk_p1[i] = packW(pm_w1 + (size_t)i * D_ * M_,  D_, M_,  M_);
        pk_p2[i] = packW(pm_w2 + (size_t)i * M_ * M_,  M_, M_,  M_);
        pk_p3[i] = packW(pm_w3 + (size_t)i * M_ * D_,  M_, D_,  D_);
    }
    bf16_t* pk_h1 = packW(hd_w1, D_, M_,   M_);
    bf16_t* pk_h2 = packW(hd_w2, M_, OUT_, 192);   // pad 151 -> 192 cols

    // ---- launch helpers ----------------------------------------------------
    auto gemm = [&](const bf16_t* A, const bf16_t* P, const float* bias,
                    const float* resid, float* Cf, bf16_t* Cb,
                    int Nr, int K, int Ncols, int Npad, int act) {
        dim3 grid((Nr + 255) / 256, Npad / 64);
        int nKT = K / 32;
        int kn  = nKT < 16 ? nKT : 16;
        size_t shmem = (size_t)kn * 4096;   // B panel chunk (4 n-tiles x kn k-tiles)
        k_gemm<<<grid, 256, shmem, stream>>>(A, P, bias, resid, Cf, Cb, Nr, K, Ncols, act);
    };
    auto ln = [&](const float* x, const float* g, const float* b, bf16_t* o, int Nr) {
        k_ln<<<dim3((Nr + 7) / 8), 256, 0, stream>>>(x, g, b, o, Nr);
    };
    const unsigned gridE = (unsigned)(((size_t)NT_ * D_ + 255) / 256);

    // ---- main L-layer pipeline --------------------------------------------
    const float* cur[5] = { img0, img1, p0, p1, p2 };

    for (int i = 0; i < L_; ++i) {
        // 5 encoder layers (image weights e=0 for streams 0-1, point e=1 for 2-4)
        for (int s = 0; s < 5; ++s) {
            int e = s < 2 ? 0 : 1;
            size_t li = (size_t)(e * L_ + i);
            ln(cur[s], enc_ln1g + li * D_, enc_ln1b + li * D_, hbf, NT_);
            gemm(hbf, pk_qkv[e][i], nullptr, nullptr, qkvf, nullptr, NT_, D_, 768, 768, 0);
            k_attn<<<dim3(B_ * H_), 64, 0, stream>>>(qkvf, obf);
            gemm(obf, pk_wo[e][i], enc_bo + li * D_, cur[s], tok[s], nullptr, NT_, D_, D_, D_, 0);
            ln(tok[s], enc_ln2g + li * D_, enc_ln2b + li * D_, hbf, NT_);
            gemm(hbf,  pk_f1[e][i], enc_b1 + li * M_, nullptr, nullptr, xbf1, NT_, D_, M_, M_, 1);
            gemm(xbf1, pk_f2[e][i], enc_b2 + li * D_, tok[s], tok[s], nullptr, NT_, M_, D_, D_, 0);
        }
        // per-stream score net (shared per-depth weights)
        for (int s = 0; s < 5; ++s) {
            ln(tok[s], sc_lng + (size_t)i * D_, sc_lnb + (size_t)i * D_, hbf, NT_);
            gemm(hbf, pk_s1[i], sc_b1 + (size_t)i * D_,  nullptr, nullptr, obf, NT_, D_, D_,  D_,  1);
            gemm(obf, pk_s2[i], sc_b2 + (size_t)i * 128, nullptr, h2f, nullptr, NT_, D_, 128, 128, 1);
            k_score<<<dim3((NT_ + 255) / 256), 256, 0, stream>>>(
                h2f, sc_w3 + (size_t)i * 128 * 2, sc_b3 + (size_t)i * 2,
                scoresb, pred, i, s);
        }
        // image token fusion
        k_fuse_img<<<dim3(gridE), 256, 0, stream>>>(tok[0], tok[1], scoresb, feat[0], feat[1]);
        // point streams: gather + patch MLP (x2) + fusion
        for (int k = 0; k < 3; ++k) {
            k_gather<<<dim3(gridE), 256, 0, stream>>>(tok[0], idx + (size_t)(k * 2 + 0) * B_ * T_, g0bf);
            k_gather<<<dim3(gridE), 256, 0, stream>>>(tok[1], idx + (size_t)(k * 2 + 1) * B_ * T_, g1bf);
            gemm(g0bf, pk_p1[i], pm_b1 + (size_t)i * M_, nullptr, nullptr, xbf1, NT_, D_, M_, M_, 1);
            gemm(xbf1, pk_p2[i], pm_b2 + (size_t)i * M_, nullptr, nullptr, xbf2, NT_, M_, M_, M_, 1);
            gemm(xbf2, pk_p3[i], pm_b3 + (size_t)i * D_, nullptr, patchA, nullptr, NT_, M_, D_, D_, 0);
            gemm(g1bf, pk_p1[i], pm_b1 + (size_t)i * M_, nullptr, nullptr, xbf1, NT_, D_, M_, M_, 1);
            gemm(xbf1, pk_p2[i], pm_b2 + (size_t)i * M_, nullptr, nullptr, xbf2, NT_, M_, M_, M_, 1);
            gemm(xbf2, pk_p3[i], pm_b3 + (size_t)i * D_, patchA, patchB, nullptr, NT_, M_, D_, D_, 0);
            k_fuse_pt<<<dim3(gridE), 256, 0, stream>>>(tok[2 + k], patchB, scoresb, 2 + k, feat[2 + k]);
        }
        for (int s = 0; s < 5; ++s) cur[s] = feat[s];
    }

    // ---- head --------------------------------------------------------------
    k_pool<<<dim3((B_ * D_ + 255) / 256), 256, 0, stream>>>(
        cur[0], cur[1], cur[2], cur[3], cur[4], pooled);
    ln(pooled, hd_lng, hd_lnb, hbf, B_);
    gemm(hbf,  pk_h1, hd_b1, nullptr, nullptr, xbf1, B_, D_, M_,   M_,  1);
    gemm(xbf1, pk_h2, hd_b2, nullptr, smplx,  nullptr, B_, M_, OUT_, 192, 0);
}
